// plainGRU_72945724555712
// MI455X (gfx1250) — matrix operand order repaired and checked by reading, not compiled, not run on hardware
//
#include <hip/hip_runtime.h>

// Problem constants (from reference): B=32, T=2048, D=1024, H=1024
#define Bb   32
#define Tt   2048
#define Dd   1024
#define Hh   1024
#define NWG  64              // persistent workgroups = hidden tiles (H/16)
#define WAVES 8              // waves per workgroup (wave32), K split 8 ways
#define BLOCK (WAVES * 32)   // 256 threads
#define KCH  (Dd / WAVES)    // 128 K-values per wave per phase (4 iters of 32)

// LDS layout: [2 phases][3 gates][16 rows][1024 K] bf16  = 192 KB weights
//             + [8 waves][8 accs][32 lanes][8 v] f32     =  64 KB partials
#define W_ELEMS   (2 * 3 * 16 * 1024)        // 98304 bf16
#define W_BYTES   (W_ELEMS * 2)              // 196608
#define P_FLOATS  (WAVES * 8 * 32 * 8)       // 16384 f32
#define SMEM_BYTES (W_BYTES + P_FLOATS * 4)  // 262144 = 256 KB

typedef __attribute__((ext_vector_type(16))) __bf16 v16bf;
typedef __attribute__((ext_vector_type(8)))  float  v8f;

union Frag {
    unsigned u[8];
    v16bf    v;
};

__device__ __forceinline__ unsigned pack2_bf16(float lo, float hi) {
    unsigned a = __float_as_uint(lo), b = __float_as_uint(hi);
    a = (a + 0x7FFFu + ((a >> 16) & 1u)) >> 16;
    b = (b + 0x7FFFu + ((b >> 16) & 1u)) >> 16;
    return (a & 0xFFFFu) | (b << 16);
}

__device__ __forceinline__ unsigned short f32_to_bf16(float f) {
    unsigned u = __float_as_uint(f);
    return (unsigned short)((u + 0x7FFFu + ((u >> 16) & 1u)) >> 16);
}

// 16x32 bf16 A-fragment from a bf16 row (K contiguous).
// lanes 0-15:  K=k0+0..7 (v0..3), k0+16..23 (v4..7); lanes 16-31: +8 / +24
__device__ __forceinline__ Frag load_a_bf16(const unsigned short* __restrict__ row, int k0, int lane) {
    const int koff = k0 + ((lane >> 4) << 3);
    uint4 a = *reinterpret_cast<const uint4*>(row + koff);
    uint4 b = *reinterpret_cast<const uint4*>(row + koff + 16);
    Frag f;
    f.u[0] = a.x; f.u[1] = a.y; f.u[2] = a.z; f.u[3] = a.w;
    f.u[4] = b.x; f.u[5] = b.y; f.u[6] = b.z; f.u[7] = b.w;
    return f;
}

// Same A layout from an f32 row with on-the-fly RNE (fallback path only).
__device__ __forceinline__ Frag load_a_f32(const float* __restrict__ row, int k0, int lane) {
    const int koff = k0 + ((lane >> 4) << 3);
    float4 q0 = *reinterpret_cast<const float4*>(row + koff);
    float4 q1 = *reinterpret_cast<const float4*>(row + koff + 4);
    float4 q2 = *reinterpret_cast<const float4*>(row + koff + 16);
    float4 q3 = *reinterpret_cast<const float4*>(row + koff + 20);
    Frag f;
    f.u[0] = pack2_bf16(q0.x, q0.y); f.u[1] = pack2_bf16(q0.z, q0.w);
    f.u[2] = pack2_bf16(q1.x, q1.y); f.u[3] = pack2_bf16(q1.z, q1.w);
    f.u[4] = pack2_bf16(q2.x, q2.y); f.u[5] = pack2_bf16(q2.z, q2.w);
    f.u[6] = pack2_bf16(q3.x, q3.y); f.u[7] = pack2_bf16(q3.z, q3.w);
    return f;
}

// 32x16 bf16 B-fragment from LDS weight slice (phase blk, gate g).
// lanes 0-15: K=0..7 (v0..3), 16..23 (v4..7); lanes 16-31: +8 / +24 (matches A).
__device__ __forceinline__ Frag load_b_lds(const unsigned short* wl, int blk, int g, int k0, int lane) {
    const int base = (((blk * 3 + g) * 16 + (lane & 15)) << 10) + k0 + ((lane >> 4) << 3);
    uint4 a = *reinterpret_cast<const uint4*>(wl + base);
    uint4 b = *reinterpret_cast<const uint4*>(wl + base + 16);
    Frag f;
    f.u[0] = a.x; f.u[1] = a.y; f.u[2] = a.z; f.u[3] = a.w;
    f.u[4] = b.x; f.u[5] = b.y; f.u[6] = b.z; f.u[7] = b.w;
    return f;
}

__device__ __forceinline__ v8f wmma_bf16(const Frag& a, const Frag& b, v8f c) {
    return __builtin_amdgcn_wmma_f32_16x16x32_bf16(false, a.v, false, b.v,
                                                   (short)0, c, false, false);
}

__device__ __forceinline__ float sigmoidf_(float x) {
    return 1.0f / (1.0f + __expf(-x));
}

// ---------------- helper kernels ----------------

__global__ void cvt_bf16x4_kernel(const float4* __restrict__ src,
                                  uint2* __restrict__ dst, int n4) {
    int i = blockIdx.x * blockDim.x + threadIdx.x;
    if (i < n4) {
        float4 q = src[i];
        dst[i] = make_uint2(pack2_bf16(q.x, q.y), pack2_bf16(q.z, q.w));
    }
}

__global__ void init_kernel(const float* __restrict__ h0,
                            float* __restrict__ hb0,
                            unsigned short* __restrict__ hbf0,
                            unsigned* __restrict__ cnt) {
    int i = blockIdx.x * blockDim.x + threadIdx.x;
    if (i == 0) *cnt = 0u;
    if (i < Bb * Hh) {
        float v = h0[i & (Hh - 1)];
        hb0[i]  = v;
        hbf0[i] = f32_to_bf16(v);
    }
}

// ---------------- persistent fused GRU ----------------
// One WG per hidden tile (16 units). Weights live in LDS; K split across 8 waves;
// LDS partial reduction + activation epilogue; grid barrier per timestep.
// XBF=true: x pre-converted to bf16; XBF=false: convert on the fly (small-ws fallback).

template <bool XBF>
__global__ __launch_bounds__(BLOCK)
void gru_persistent(const float* __restrict__ xf,
                    const unsigned short* __restrict__ xbf,   // bf16 [B,T,D]
                    const unsigned short* __restrict__ wtih,  // bf16 [3H, D]
                    const unsigned short* __restrict__ wthh,  // bf16 [3H, H]
                    const float* __restrict__ b_ih,
                    const float* __restrict__ b_hh,
                    float* __restrict__ hb0, float* __restrict__ hb1,
                    unsigned short* __restrict__ hbf0, unsigned short* __restrict__ hbf1,
                    unsigned* __restrict__ cnt,
                    float* __restrict__ out_seq, float* __restrict__ out_last) {
    extern __shared__ char smem[];
    unsigned short* wl = reinterpret_cast<unsigned short*>(smem);
    float*          pl = reinterpret_cast<float*>(smem + W_BYTES);

    const int tid  = threadIdx.x;
    const int lane = tid & 31;
    const int wv   = tid >> 5;                  // wave id 0..7 = K chunk
    const int col0 = blockIdx.x << 4;           // tile: hidden units col0..col0+15
    const int nn   = lane & 15;
    const int j    = col0 + nn;                 // this lane's hidden unit

    // ---- stage this tile's weights into LDS (one time) ----
    // flat element index e = R*1024 + k, R = (blk*3+g)*16 + r
    for (int vi = tid; vi < W_ELEMS / 8; vi += BLOCK) {
        const int e = vi << 3;
        const int R = e >> 10;                  // 0..95
        const int k = e & 1023;
        const int blk = R / 48;
        const int rem = R - blk * 48;
        const int g = rem >> 4;
        const int r = rem & 15;
        const unsigned short* src = (blk ? wthh : wtih)
                                  + (size_t)((g << 10) + col0 + r) * 1024 + k;
        *reinterpret_cast<uint4*>(wl + e) = *reinterpret_cast<const uint4*>(src);
    }
    __syncthreads();

    const float br  = b_ih[j] + b_hh[j];
    const float bz  = b_ih[Hh + j] + b_hh[Hh + j];
    const float bxn = b_ih[2 * Hh + j];
    const float bhn = b_hh[2 * Hh + j];

    const float*          xr0f = xf  + (size_t)nn * (size_t)Tt * Dd;
    const float*          xr1f = xf  + (size_t)(nn + 16) * (size_t)Tt * Dd;
    const unsigned short* xr0b = xbf + (size_t)nn * (size_t)Tt * Dd;
    const unsigned short* xr1b = xbf + (size_t)(nn + 16) * (size_t)Tt * Dd;

    // epilogue role: thread handles 2 elements (mt_e, v = vb, vb+1) at its lane
    const int mt_e = wv & 1;
    const int vb   = (wv >> 1) << 1;

    const int kw = wv * KCH;                    // this wave's K chunk base

    #pragma unroll 1
    for (int t = 0; t < Tt; ++t) {
        const float*          hc   = (t & 1) ? hb1  : hb0;
        float*                hnx  = (t & 1) ? hb0  : hb1;
        const unsigned short* hcb  = (t & 1) ? hbf1 : hbf0;
        unsigned short*       hnxb = (t & 1) ? hbf0 : hbf1;

        v8f aR0 = {}, aR1 = {}, aZ0 = {}, aZ1 = {};
        v8f aNx0 = {}, aNx1 = {}, aNh0 = {}, aNh1 = {};

        // ---- input-side GEMM chunk: x_t . W_ih^T, K in [kw, kw+KCH) ----
        #pragma unroll
        for (int ki = 0; ki < KCH; ki += 32) {
            const int k0 = kw + ki;
            Frag a0, a1;
            if (XBF) {
                a0 = load_a_bf16(xr0b + (size_t)t * Dd, k0, lane);
                a1 = load_a_bf16(xr1b + (size_t)t * Dd, k0, lane);
            } else {
                a0 = load_a_f32(xr0f + (size_t)t * Dd, k0, lane);
                a1 = load_a_f32(xr1f + (size_t)t * Dd, k0, lane);
            }
            Frag fr = load_b_lds(wl, 0, 0, k0, lane);
            Frag fz = load_b_lds(wl, 0, 1, k0, lane);
            Frag fn = load_b_lds(wl, 0, 2, k0, lane);
            aR0  = wmma_bf16(a0, fr, aR0);   aR1  = wmma_bf16(a1, fr, aR1);
            aZ0  = wmma_bf16(a0, fz, aZ0);   aZ1  = wmma_bf16(a1, fz, aZ1);
            aNx0 = wmma_bf16(a0, fn, aNx0);  aNx1 = wmma_bf16(a1, fn, aNx1);
        }

        // ---- hidden-side GEMM chunk: h . W_hh^T (h already bf16) ----
        const unsigned short* hr0 = hcb + (size_t)nn * Hh;
        const unsigned short* hr1 = hcb + (size_t)(nn + 16) * Hh;
        #pragma unroll
        for (int ki = 0; ki < KCH; ki += 32) {
            const int k0 = kw + ki;
            Frag a0 = load_a_bf16(hr0, k0, lane);
            Frag a1 = load_a_bf16(hr1, k0, lane);
            Frag fr = load_b_lds(wl, 1, 0, k0, lane);
            Frag fz = load_b_lds(wl, 1, 1, k0, lane);
            Frag fn = load_b_lds(wl, 1, 2, k0, lane);
            aR0  = wmma_bf16(a0, fr, aR0);   aR1  = wmma_bf16(a1, fr, aR1);
            aZ0  = wmma_bf16(a0, fz, aZ0);   aZ1  = wmma_bf16(a1, fz, aZ1);
            aNh0 = wmma_bf16(a0, fn, aNh0);  aNh1 = wmma_bf16(a1, fn, aNh1);
        }

        // ---- store partial accumulators: pl[w][a][lane][v] ----
        {
            v8f accs[8] = { aR0, aR1, aZ0, aZ1, aNx0, aNx1, aNh0, aNh1 };
            #pragma unroll
            for (int a = 0; a < 8; ++a) {
                float* p = pl + (((wv * 8 + a) * 32 + lane) << 3);
                reinterpret_cast<float4*>(p)[0] =
                    make_float4(accs[a][0], accs[a][1], accs[a][2], accs[a][3]);
                reinterpret_cast<float4*>(p)[1] =
                    make_float4(accs[a][4], accs[a][5], accs[a][6], accs[a][7]);
            }
        }
        __syncthreads();

        // ---- reduce 8 partials x 4 gates and run activation epilogue ----
        {
            float2 sR = {0.f, 0.f}, sZ = {0.f, 0.f}, sNx = {0.f, 0.f}, sNh = {0.f, 0.f};
            #pragma unroll
            for (int w = 0; w < WAVES; ++w) {
                const int base = ((w * 8) * 32 + lane) * 8 + vb;
                float2 r2  = *reinterpret_cast<const float2*>(pl + base + ((0 * 2 + mt_e) << 8));
                float2 z2  = *reinterpret_cast<const float2*>(pl + base + ((1 * 2 + mt_e) << 8));
                float2 nx2 = *reinterpret_cast<const float2*>(pl + base + ((2 * 2 + mt_e) << 8));
                float2 nh2 = *reinterpret_cast<const float2*>(pl + base + ((3 * 2 + mt_e) << 8));
                sR.x += r2.x;   sR.y += r2.y;
                sZ.x += z2.x;   sZ.y += z2.y;
                sNx.x += nx2.x; sNx.y += nx2.y;
                sNh.x += nh2.x; sNh.y += nh2.y;
            }
            #pragma unroll
            for (int e = 0; e < 2; ++e) {
                const int v = vb + e;
                const int m = (mt_e << 4) + ((lane >> 4) << 3) + v;
                const float rs  = e ? sR.y  : sR.x;
                const float zs  = e ? sZ.y  : sZ.x;
                const float nxs = e ? sNx.y : sNx.x;
                const float nhs = e ? sNh.y : sNh.x;
                float r    = sigmoidf_(rs + br);
                float z    = sigmoidf_(zs + bz);
                float nval = tanhf(nxs + bxn + r * (nhs + bhn));
                float hold = hc[(size_t)m * Hh + j];
                float hnew = (1.0f - z) * nval + z * hold;
                hnx[(size_t)m * Hh + j]  = hnew;
                hnxb[(size_t)m * Hh + j] = f32_to_bf16(hnew);
                out_seq[((size_t)m * Tt + t) * Hh + j] = hnew;
                if (t == Tt - 1) out_last[(size_t)m * Hh + j] = hnew;
            }
        }

        // ---- grid-wide generation barrier across NWG workgroups ----
        __threadfence();
        __syncthreads();
        if (tid == 0) {
            atomicAdd(cnt, 1u);
            const unsigned target = (unsigned)NWG * (unsigned)(t + 1);
            while (__hip_atomic_load(cnt, __ATOMIC_ACQUIRE, __HIP_MEMORY_SCOPE_AGENT) < target) {
                __builtin_amdgcn_s_sleep(2);
            }
        }
        __syncthreads();
        __threadfence();
    }
}

// ---------------- launcher ----------------

extern "C" void kernel_launch(void* const* d_in, const int* in_sizes, int n_in,
                              void* d_out, int out_size, void* d_ws, size_t ws_size,
                              hipStream_t stream) {
    const float* x    = (const float*)d_in[0];
    const float* w_ih = (const float*)d_in[1];
    const float* w_hh = (const float*)d_in[2];
    const float* b_ih = (const float*)d_in[3];
    const float* b_hh = (const float*)d_in[4];
    const float* h0   = (const float*)d_in[5];

    char* ws = (char*)d_ws;
    size_t off = 0;
    unsigned* cnt = (unsigned*)(ws + off);              off += 256;
    unsigned short* wtih = (unsigned short*)(ws + off); off += (size_t)3 * Hh * Dd * 2;
    unsigned short* wthh = (unsigned short*)(ws + off); off += (size_t)3 * Hh * Hh * 2;
    float* hb0 = (float*)(ws + off);                    off += (size_t)Bb * Hh * 4;
    float* hb1 = (float*)(ws + off);                    off += (size_t)Bb * Hh * 4;
    unsigned short* hbf0 = (unsigned short*)(ws + off); off += (size_t)Bb * Hh * 2;
    unsigned short* hbf1 = (unsigned short*)(ws + off); off += (size_t)Bb * Hh * 2;
    off = (off + 255) & ~(size_t)255;
    unsigned short* xbf = (unsigned short*)(ws + off);
    const size_t need_big = off + (size_t)Bb * Tt * Dd * 2;
    const bool big = (ws_size >= need_big);

    float* out_seq  = (float*)d_out;
    float* out_last = out_seq + (size_t)Bb * Tt * Hh;

    const int nW4 = (3 * Hh * Dd) / 4;
    cvt_bf16x4_kernel<<<(nW4 + 255) / 256, 256, 0, stream>>>(
        (const float4*)w_ih, (uint2*)wtih, nW4);
    cvt_bf16x4_kernel<<<(nW4 + 255) / 256, 256, 0, stream>>>(
        (const float4*)w_hh, (uint2*)wthh, nW4);

    if (big) {
        const int nX4 = (int)(((size_t)Bb * Tt * Dd) / 4);
        cvt_bf16x4_kernel<<<(nX4 + 255) / 256, 256, 0, stream>>>(
            (const float4*)x, (uint2*)xbf, nX4);
    }

    init_kernel<<<(Bb * Hh + 255) / 256, 256, 0, stream>>>(h0, hb0, hbf0, cnt);

    if (big) {
        gru_persistent<true><<<NWG, BLOCK, SMEM_BYTES, stream>>>(
            x, xbf, wtih, wthh, b_ih, b_hh, hb0, hb1, hbf0, hbf1, cnt, out_seq, out_last);
    } else {
        gru_persistent<false><<<NWG, BLOCK, SMEM_BYTES, stream>>>(
            x, xbf, wtih, wthh, b_ih, b_hh, hb0, hb1, hbf0, hbf1, cnt, out_seq, out_last);
    }
}
